// AttentionWithKVCache_6571299963362
// MI455X (gfx1250) — matrix operand order repaired and checked by reading, not compiled, not run on hardware
//
#include <hip/hip_runtime.h>
#include <hip/hip_bf16.h>

// ---------------------------------------------------------------------------
// Causal MHA forward for MI455X (gfx1250, wave32, WMMA bf16).
//   d_out = softmax(mask((x Wq^T)(x Wk^T)^T / sqrt(Dh))) (x Wv^T) Wo^T
// Pipeline: fp32->bf16 convert, 3x bf16 GEMM (QKV, double-buffered LDS),
// flash attention (register softmax, L2 prefetch), final GEMM (fp32 out).
// ---------------------------------------------------------------------------

#define D_MODEL 2048
#define N_HEADS 16
#define HEAD_DIM 128
#define BATCH 2
#define SEQ 2048
#define ROWS (BATCH * SEQ)   // 4096

typedef __bf16 bf16;
typedef __attribute__((ext_vector_type(16))) __bf16 v16bf;
typedef __attribute__((ext_vector_type(8)))  float  v8f;
typedef __attribute__((ext_vector_type(4)))  unsigned int u32x4;

union Frag16 {            // 16 bf16 = one WMMA A/B operand (8 VGPRs)
  v16bf v;
  u32x4 u[2];
};
union Cvt8 {              // 8 bf16 <-> one 16B vector
  u32x4 u;
  bf16  e[8];
};

__device__ __forceinline__ v8f wmma_bf16(const v16bf& a, const v16bf& b, const v8f& c) {
  return __builtin_amdgcn_wmma_f32_16x16x32_bf16(false, a, false, b, (short)0, c,
                                                 false, false);
}

// ---------------------------------------------------------------------------
// fp32 -> bf16 conversion (vectorized x4)
// ---------------------------------------------------------------------------
struct bf16x4 { bf16 a, b, c, d; };

__global__ void cvt_f32_to_bf16(const float* __restrict__ src,
                                bf16* __restrict__ dst, int n4) {
  int i = blockIdx.x * blockDim.x + threadIdx.x;
  int stride = gridDim.x * blockDim.x;
  for (; i < n4; i += stride) {
    float4 f = ((const float4*)src)[i];
    bf16x4 o;
    o.a = (bf16)f.x; o.b = (bf16)f.y; o.c = (bf16)f.z; o.d = (bf16)f.w;
    ((bf16x4*)dst)[i] = o;
  }
}

// ---------------------------------------------------------------------------
// GEMM: Y[M,N] = A[M,K] * W[N,K]^T   (both row-major along K -> "NT")
// 128x128 tile / 256 threads (8 waves); wave = 32x64 (2x4 WMMA frags).
// Double-buffered LDS; all 6 fragments preloaded into distinct registers
// before the 8-WMMA burst so ds waits stagger instead of full drains.
// Writes bf16 (outB) or fp32 (outF).
// ---------------------------------------------------------------------------
#define GT_M 128
#define GT_N 128
#define GT_K 64
#define GK_PAD 8           // LDS row stride 72 elems = 144B (16B aligned)

__global__ __launch_bounds__(256)
void gemm_bf16_nt(const bf16* __restrict__ A, const bf16* __restrict__ W,
                  bf16* __restrict__ outB, float* __restrict__ outF,
                  int M, int N, int K) {
  __shared__ bf16 As[2][GT_M][GT_K + GK_PAD];
  __shared__ bf16 Bs[2][GT_N][GT_K + GK_PAD];

  const int tid  = threadIdx.x;
  const int lane = tid & 31;
  const int wid  = tid >> 5;
  const int wm   = wid & 3;      // wave m-offset /32
  const int wn   = wid >> 2;     // wave n-offset /64
  const int mbase = blockIdx.y * GT_M;
  const int nbase = blockIdx.x * GT_N;

  const int l15 = lane & 15;
  const int hi  = (lane >= 16);

  v8f acc[2][4];
#pragma unroll
  for (int mf = 0; mf < 2; ++mf)
#pragma unroll
    for (int nf = 0; nf < 4; ++nf) acc[mf][nf] = (v8f)0.0f;

  const int lrow   = tid >> 3;        // 0..31
  const int lchunk = (tid & 7) * 8;   // 0..56 (elems)

  // ---- prologue: stage K-tile 0 into LDS buffer 0 ----
  u32x4 pa[4], pb[4];
#pragma unroll
  for (int r = 0; r < 4; ++r) {
    pa[r] = *(const u32x4*)&A[(size_t)(mbase + lrow + r * 32) * K + lchunk];
    pb[r] = *(const u32x4*)&W[(size_t)(nbase + lrow + r * 32) * K + lchunk];
  }
#pragma unroll
  for (int r = 0; r < 4; ++r) {
    *(u32x4*)&As[0][lrow + r * 32][lchunk] = pa[r];
    *(u32x4*)&Bs[0][lrow + r * 32][lchunk] = pb[r];
  }
  __syncthreads();

  int buf = 0;
  for (int k0 = 0; k0 < K; k0 += GT_K) {
    const int knext = k0 + GT_K;
    const bool more = knext < K;

    // ---- prefetch next K-tile into registers (overlaps with WMMA below) ----
    if (more) {
#pragma unroll
      for (int r = 0; r < 4; ++r) {
        pa[r] = *(const u32x4*)&A[(size_t)(mbase + lrow + r * 32) * K + knext + lchunk];
        pb[r] = *(const u32x4*)&W[(size_t)(nbase + lrow + r * 32) * K + knext + lchunk];
      }
    }

    // ---- compute from LDS buffer `buf` ----
#pragma unroll
    for (int kk = 0; kk < GT_K; kk += 32) {
      // Preload ALL fragments for this k-step into distinct registers:
      // 12 ds_load_b128 in flight, then 8 back-to-back WMMAs.
      Frag16 af[2], bfr[4];
#pragma unroll
      for (int mf = 0; mf < 2; ++mf) {
        // A frag: lanes 0-15 -> K {kk..kk+7, kk+16..kk+23}; lanes 16-31 +8
        const int arow = wm * 32 + mf * 16 + l15;
        const int ako  = kk + (hi ? 8 : 0);
        af[mf].u[0] = *(const u32x4*)&As[buf][arow][ako];
        af[mf].u[1] = *(const u32x4*)&As[buf][arow][ako + 16];
      }
#pragma unroll
      for (int nf = 0; nf < 4; ++nf) {
        // B frag: lanes 0-15 -> K {kk..kk+7, kk+16..kk+23}; lanes 16-31 +8
        const int brow = wn * 64 + nf * 16 + l15;
        const int bko  = kk + (hi ? 8 : 0);
        bfr[nf].u[0] = *(const u32x4*)&Bs[buf][brow][bko];
        bfr[nf].u[1] = *(const u32x4*)&Bs[buf][brow][bko + 16];
      }
#pragma unroll
      for (int nf = 0; nf < 4; ++nf)
#pragma unroll
        for (int mf = 0; mf < 2; ++mf)
          acc[mf][nf] = wmma_bf16(af[mf].v, bfr[nf].v, acc[mf][nf]);
    }

    // ---- store prefetched tile into the other buffer; one barrier ----
    if (more) {
      // safe: previous iteration's barrier guarantees buf^1 is drained
#pragma unroll
      for (int r = 0; r < 4; ++r) {
        *(u32x4*)&As[buf ^ 1][lrow + r * 32][lchunk] = pa[r];
        *(u32x4*)&Bs[buf ^ 1][lrow + r * 32][lchunk] = pb[r];
      }
      __syncthreads();
      buf ^= 1;
    }
  }

  // Epilogue. C/D layout: lane<16: M=r,N=lane ; lane>=16: M=8+r,N=lane-16.
  const int mo = mbase + wm * 32 + (hi ? 8 : 0);
  const int no = nbase + wn * 64 + l15;
#pragma unroll
  for (int mf = 0; mf < 2; ++mf)
#pragma unroll
    for (int nf = 0; nf < 4; ++nf)
#pragma unroll
      for (int r = 0; r < 8; ++r) {
        const size_t idx = (size_t)(mo + mf * 16 + r) * N + (no + nf * 16);
        if (outF) outF[idx] = acc[mf][nf][r];
        else      outB[idx] = (bf16)acc[mf][nf][r];
      }
}

// ---------------------------------------------------------------------------
// Flash attention (causal). One WG = (b, h, 128-query tile), 256 thr, 8 waves.
// Each wave owns 16 query rows. Key blocks of 64. Online softmax in registers.
// Next K/V block is prefetched into L2/WGP$ while the current one computes.
// ---------------------------------------------------------------------------
#define QT 128
#define KT 64

__global__ __launch_bounds__(256)
void flash_attn_causal(const bf16* __restrict__ Q, const bf16* __restrict__ K,
                       const bf16* __restrict__ V, bf16* __restrict__ O) {
  __shared__ bf16 Qs[QT][HEAD_DIM + 8];       // stride 136
  __shared__ bf16 Ks[KT][HEAD_DIM + 8];       // stride 136
  __shared__ bf16 Vt[HEAD_DIM][KT + 8];       // V transposed, stride 72
  __shared__ bf16 Ps[QT][KT + 8];             // probs bf16, stride 72

  const int qtile = blockIdx.x;
  const int h     = blockIdx.y;
  const int b     = blockIdx.z;
  const int qb    = qtile * QT;

  const int tid  = threadIdx.x;
  const int lane = tid & 31;
  const int wid  = tid >> 5;
  const int l15  = lane & 15;
  const int hi   = (lane >= 16);
  const int hi8  = hi ? 8 : 0;

  const size_t hcol = (size_t)h * HEAD_DIM;
  const size_t rb   = (size_t)b * SEQ;

  const int lrow   = tid >> 4;        // 0..15
  const int lchunk = (tid & 15) * 8;  // 0..120

  // ---- load Q tile (reused for the whole kernel) ----
#pragma unroll
  for (int r = 0; r < QT; r += 16)
    *(u32x4*)&Qs[lrow + r][lchunk] =
        *(const u32x4*)&Q[(rb + qb + lrow + r) * D_MODEL + hcol + lchunk];
  __syncthreads();

  // ---- preload Q A-fragments for this wave's 16 rows (4 k-steps of 32) ----
  Frag16 qa[4];
  const int arow = wid * 16 + l15;
  const int aoff = hi ? 8 : 0;
#pragma unroll
  for (int ks = 0; ks < 4; ++ks) {
    qa[ks].u[0] = *(const u32x4*)&Qs[arow][ks * 32 + aoff];
    qa[ks].u[1] = *(const u32x4*)&Qs[arow][ks * 32 + aoff + 16];
  }

  v8f o[8];
#pragma unroll
  for (int nf = 0; nf < 8; ++nf) o[nf] = (v8f)0.0f;
  float mrun[8], lrun[8];
#pragma unroll
  for (int r = 0; r < 8; ++r) { mrun[r] = -INFINITY; lrun[r] = 0.0f; }

  const float scale = 0.08838834764831845f;   // 1/sqrt(128)
  const int kend = qb + QT;                   // causal: keys < qb+128

  for (int kb = 0; kb < kend; kb += KT) {
    __syncthreads();   // previous iteration's use of Ks/Vt is complete

    // ---- stage K block (row-major) and V block (transposed) ----
#pragma unroll
    for (int r = 0; r < KT; r += 16) {
      *(u32x4*)&Ks[lrow + r][lchunk] =
          *(const u32x4*)&K[(rb + kb + lrow + r) * D_MODEL + hcol + lchunk];
      Cvt8 cv;
      cv.u = *(const u32x4*)&V[(rb + kb + lrow + r) * D_MODEL + hcol + lchunk];
#pragma unroll
      for (int i = 0; i < 8; ++i) Vt[lchunk + i][lrow + r] = cv.e[i];
    }

    // ---- warm L2/WGP$ with the next K/V block (global_prefetch_b8) ----
    if (kb + KT < kend) {
      __builtin_prefetch(&K[(rb + kb + KT + lrow) * D_MODEL + hcol + lchunk], 0, 1);
      __builtin_prefetch(&V[(rb + kb + KT + lrow) * D_MODEL + hcol + lchunk], 0, 1);
    }
    __syncthreads();

    // ---- S = Q K^T (16 x 64 per wave) ----
    v8f s[4];
#pragma unroll
    for (int nf = 0; nf < 4; ++nf) s[nf] = (v8f)0.0f;
#pragma unroll
    for (int ks = 0; ks < 4; ++ks) {
#pragma unroll
      for (int nf = 0; nf < 4; ++nf) {
        Frag16 bk;
        const int brow = nf * 16 + l15;            // key within block
        const int bko  = ks * 32 + (hi ? 8 : 0);   // dh offset
        bk.u[0] = *(const u32x4*)&Ks[brow][bko];
        bk.u[1] = *(const u32x4*)&Ks[brow][bko + 16];
        s[nf] = wmma_bf16(qa[ks].v, bk.v, s[nf]);
      }
    }

    // ---- scale + causal mask (elementwise in accumulator layout) ----
    const int qrow0 = qb + wid * 16 + hi8;    // + r
    const int kcol0 = kb + l15;               // + nf*16
#pragma unroll
    for (int nf = 0; nf < 4; ++nf)
#pragma unroll
      for (int r = 0; r < 8; ++r) {
        float v = s[nf][r] * scale;
        if (kcol0 + nf * 16 > qrow0 + r) v = -INFINITY;
        s[nf][r] = v;
      }

    // ---- row max across 64 keys: frags in-register + 16-lane shfl tree ----
    float rm[8];
#pragma unroll
    for (int r = 0; r < 8; ++r) {
      rm[r] = fmaxf(fmaxf(s[0][r], s[1][r]), fmaxf(s[2][r], s[3][r]));
#pragma unroll
      for (int m = 1; m < 16; m <<= 1) rm[r] = fmaxf(rm[r], __shfl_xor(rm[r], m, 32));
    }
    float alpha[8];
#pragma unroll
    for (int r = 0; r < 8; ++r) {
      const float mnew = fmaxf(mrun[r], rm[r]);
      alpha[r] = __expf(mrun[r] - mnew);
      mrun[r] = mnew;
    }

    // ---- probabilities + row sum ----
    float rs[8];
#pragma unroll
    for (int r = 0; r < 8; ++r) rs[r] = 0.0f;
#pragma unroll
    for (int nf = 0; nf < 4; ++nf)
#pragma unroll
      for (int r = 0; r < 8; ++r) {
        const float p = __expf(s[nf][r] - mrun[r]);
        s[nf][r] = p;
        rs[r] += p;
      }
#pragma unroll
    for (int r = 0; r < 8; ++r) {
#pragma unroll
      for (int m = 1; m < 16; m <<= 1) rs[r] += __shfl_xor(rs[r], m, 32);
      lrun[r] = alpha[r] * lrun[r] + rs[r];
    }

    // ---- rescale O, write P (per-wave LDS slab, same-wave ordering) ----
#pragma unroll
    for (int nf = 0; nf < 8; ++nf)
#pragma unroll
      for (int r = 0; r < 8; ++r) o[nf][r] *= alpha[r];
    const int prow = wid * 16 + hi8;
#pragma unroll
    for (int nf = 0; nf < 4; ++nf)
#pragma unroll
      for (int r = 0; r < 8; ++r)
        Ps[prow + r][nf * 16 + l15] = (bf16)s[nf][r];

    // ---- O += P V  (16 x 128 per wave) ----
#pragma unroll
    for (int ks = 0; ks < 2; ++ks) {
      Frag16 pa;
      pa.u[0] = *(const u32x4*)&Ps[arow][ks * 32 + aoff];
      pa.u[1] = *(const u32x4*)&Ps[arow][ks * 32 + aoff + 16];
#pragma unroll
      for (int nf = 0; nf < 8; ++nf) {
        Frag16 bv;
        const int brow = nf * 16 + l15;            // dh
        const int bko  = ks * 32 + (hi ? 8 : 0);   // key
        bv.u[0] = *(const u32x4*)&Vt[brow][bko];
        bv.u[1] = *(const u32x4*)&Vt[brow][bko + 16];
        o[nf] = wmma_bf16(pa.v, bv.v, o[nf]);
      }
    }
  }

  // ---- normalize and store bf16 attention output ----
#pragma unroll
  for (int nf = 0; nf < 8; ++nf)
#pragma unroll
    for (int r = 0; r < 8; ++r) {
      const float v = o[nf][r] / lrun[r];
      const size_t row = rb + qb + wid * 16 + hi8 + r;
      const size_t col = hcol + nf * 16 + l15;
      O[row * D_MODEL + col] = (bf16)v;
    }
}

// ---------------------------------------------------------------------------
// Host-side launch
// ---------------------------------------------------------------------------
extern "C" void kernel_launch(void* const* d_in, const int* in_sizes, int n_in,
                              void* d_out, int out_size, void* d_ws, size_t ws_size,
                              hipStream_t stream) {
  (void)in_sizes; (void)n_in; (void)out_size; (void)ws_size;

  const float* x  = (const float*)d_in[0];
  const float* wq = (const float*)d_in[1];
  const float* wk = (const float*)d_in[2];
  const float* wv = (const float*)d_in[3];
  const float* wo = (const float*)d_in[4];

  const size_t NX = (size_t)ROWS * D_MODEL;      // 8388608
  const size_t NW = (size_t)D_MODEL * D_MODEL;   // 4194304

  bf16* xb  = (bf16*)d_ws;
  bf16* wqb = xb  + NX;
  bf16* wkb = wqb + NW;
  bf16* wvb = wkb + NW;
  bf16* wob = wvb + NW;
  bf16* Qb  = wob + NW;
  bf16* Kb  = Qb  + NX;
  bf16* Vb  = Kb  + NX;
  bf16* Ab  = Vb  + NX;

  // fp32 -> bf16
  cvt_f32_to_bf16<<<2048, 256, 0, stream>>>(x,  xb,  (int)(NX / 4));
  cvt_f32_to_bf16<<<1024, 256, 0, stream>>>(wq, wqb, (int)(NW / 4));
  cvt_f32_to_bf16<<<1024, 256, 0, stream>>>(wk, wkb, (int)(NW / 4));
  cvt_f32_to_bf16<<<1024, 256, 0, stream>>>(wv, wvb, (int)(NW / 4));
  cvt_f32_to_bf16<<<1024, 256, 0, stream>>>(wo, wob, (int)(NW / 4));

  // Q/K/V projections: (4096 x 2048) = x * W^T
  dim3 gGemm(D_MODEL / GT_N, ROWS / GT_M);       // (16, 32)
  gemm_bf16_nt<<<gGemm, 256, 0, stream>>>(xb, wqb, Qb, nullptr, ROWS, D_MODEL, D_MODEL);
  gemm_bf16_nt<<<gGemm, 256, 0, stream>>>(xb, wkb, Kb, nullptr, ROWS, D_MODEL, D_MODEL);
  gemm_bf16_nt<<<gGemm, 256, 0, stream>>>(xb, wvb, Vb, nullptr, ROWS, D_MODEL, D_MODEL);

  // causal flash attention per (qtile, head, batch)
  dim3 gAttn(SEQ / QT, N_HEADS, BATCH);          // (16, 16, 2)
  flash_attn_causal<<<gAttn, 256, 0, stream>>>(Qb, Kb, Vb, Ab);

  // output projection: fp32 result
  gemm_bf16_nt<<<gGemm, 256, 0, stream>>>(Ab, wob, nullptr, (float*)d_out,
                                          ROWS, D_MODEL, D_MODEL);
}
